// MultiheadAttentionMTG_88699664597589
// MI455X (gfx1250) — compile-verified, matrix operand order checked by reading, not run on hardware
//
#include <hip/hip_runtime.h>
#include <stdint.h>

#define T_DIM 2048
#define B_DIM 2
#define E_DIM 1024
#define H_DIM 16
#define HD_DIM 64

typedef __attribute__((ext_vector_type(16))) __bf16 v16bf;
typedef __attribute__((ext_vector_type(8)))  __bf16 v8bf;
typedef __attribute__((ext_vector_type(4)))  __bf16 v4bf;
typedef __attribute__((ext_vector_type(8)))  float  v8f;
typedef __attribute__((ext_vector_type(4)))  float  vf4;
typedef __attribute__((ext_vector_type(4)))  uint32_t u32x4;
typedef __attribute__((ext_vector_type(8)))  uint32_t u32x8;

// ---- CDNA5 async global->LDS copy (ASYNCcnt-tracked, no VGPR round trip) ----
static __device__ __forceinline__ void async_copy16(uint32_t lds_off, const void* gptr)
{
    asm volatile("global_load_async_to_lds_b128 %0, %1, off"
                 :: "v"(lds_off), "v"((uint64_t)(uintptr_t)gptr)
                 : "memory");
}
static __device__ __forceinline__ void wait_async0()
{
    asm volatile("s_wait_asynccnt 0x0" ::: "memory");
}

// ---- CDNA5 Tensor Data Mover: 1-D tile load, bf16 elements ----
// D# group0: [1:0]=count=1, [63:32]=lds_addr, [120:64]=global_addr, [127:126]=type=2
// D# group1: [17:16]=data_size(1 -> 2B), tensor_dim0=n, tensor_dim1=1,
//            tile_dim0=n (<=65535), tile_dim1=0 (unused -> 1-D), stride0=n.
static __device__ __forceinline__ void tdm_load_bf16(uint32_t lds_off, const void* gptr,
                                                     uint32_t n /* elements (2B) */)
{
    uint64_t ga = (uint64_t)(uintptr_t)gptr;
    uint32_t g0w1 = __builtin_amdgcn_readfirstlane(lds_off);
    uint32_t g0w2 = __builtin_amdgcn_readfirstlane((uint32_t)ga);
    uint32_t g0w3 = __builtin_amdgcn_readfirstlane(
        (uint32_t)((ga >> 32) & 0x1FFFFFFu) | (2u << 30));
    uint32_t nlo16 = __builtin_amdgcn_readfirstlane(n << 16);

    u32x4 g0 = { 1u, g0w1, g0w2, g0w3 };
    u32x8 g1 = { (1u << 16),          // data_size = 2 bytes
                 nlo16,               // tensor_dim0[15:0] << 16
                 (n >> 16) | (1u << 16), // tensor_dim0[31:16] | tensor_dim1=1
                 nlo16,               // tile_dim0 << 16
                 0u,                  // tile_dim1=0 (unused), tile_dim2=0
                 n,                   // tensor_dim0_stride lo
                 0u, 0u };
    asm volatile("tensor_load_to_lds %0, %1" :: "s"(g0), "s"(g1) : "memory");
}
static __device__ __forceinline__ void wait_tensor0()
{
    __builtin_amdgcn_s_wait_tensorcnt(0);
}

// Load a 16-element bf16 WMMA operand fragment (A: 16x32 row / B: 32x16 col).
// rowbase points at this lane's 32-element K-chunk. Fragment element -> K:
//   e in [0,8):  K = kb + e            (kb = (lane>>4)*8)
//   e in [8,16): K = 16 + kb + (e-8)
static __device__ __forceinline__ v16bf load_frag32(const __bf16* rowbase, int kb)
{
    v4bf a = *(const v4bf*)(rowbase + kb);
    v4bf b = *(const v4bf*)(rowbase + kb + 4);
    v4bf c = *(const v4bf*)(rowbase + kb + 16);
    v4bf d = *(const v4bf*)(rowbase + kb + 20);
    v16bf f;
#pragma unroll
    for (int i = 0; i < 4; ++i) { f[i] = a[i]; f[4+i] = b[i]; f[8+i] = c[i]; f[12+i] = d[i]; }
    return f;
}

// Column gather (stride HD) for V^T fragments from row-major V tile in LDS.
static __device__ __forceinline__ v16bf load_frag_col(const __bf16* col, int kb)
{
    v16bf f;
#pragma unroll
    for (int e = 0; e < 8; ++e) f[e]     = col[(kb + e) * HD_DIM];
#pragma unroll
    for (int e = 0; e < 8; ++e) f[8 + e] = col[(16 + kb + e) * HD_DIM];
    return f;
}

static __device__ __forceinline__ v8f wmma_bf16(v16bf a, v16bf b, v8f c)
{
    return __builtin_amdgcn_wmma_f32_16x16x32_bf16(false, a, false, b, (short)0, c, false, false);
}

// ---------------- Kernel 1: fused Q/K/V projection ----------------
// Y = X @ W^T + bias ; write bf16 into [B,H,T,HD]; fold 1/sqrt(HD) into Q.
__global__ __launch_bounds__(128)
void qkv_proj_kernel(const float* __restrict__ xq, const float* __restrict__ xk,
                     const float* __restrict__ xv,
                     const float* __restrict__ Wq, const float* __restrict__ bq,
                     const float* __restrict__ Wk, const float* __restrict__ bk,
                     const float* __restrict__ Wv, const float* __restrict__ bv,
                     __bf16* __restrict__ qo, __bf16* __restrict__ ko,
                     __bf16* __restrict__ vo)
{
    __shared__ __bf16 At[64 * 32];   // [m_local][k_local]
    __shared__ __bf16 Bt[64 * 32];   // [n_local][k_local]  (rows of W)

    const int z = blockIdx.z;
    const float* X    = (z == 0) ? xq : (z == 1) ? xk : xv;
    const float* W    = (z == 0) ? Wq : (z == 1) ? Wk : Wv;
    const float* bias = (z == 0) ? bq : (z == 1) ? bk : bv;
    __bf16* dst       = (z == 0) ? qo : (z == 1) ? ko : vo;
    const float scale = (z == 0) ? 0.125f : 1.0f;   // 1/sqrt(64) folded into Q

    const int tid  = threadIdx.x;
    const int wave = tid >> 5;
    const int lane = tid & 31;
    const int M0   = blockIdx.x * 64;
    const int N0   = blockIdx.y * 64;
    const int kb   = (lane >> 4) << 3;

    v8f acc[4] = {};

    for (int k0 = 0; k0 < E_DIM; k0 += 32) {
        __syncthreads();
        // vectorized convert+stage: 8 elems per chunk, 2 chunks per thread/tile
        for (int i = tid; i < 256; i += 128) {
            int r = i >> 2, c = (i & 3) << 3;
            const float* xa = X + (size_t)(M0 + r) * E_DIM + k0 + c;
            const float* wa = W + (size_t)(N0 + r) * E_DIM + k0 + c;
            vf4 x0 = *(const vf4*)xa, x1 = *(const vf4*)(xa + 4);
            vf4 w0 = *(const vf4*)wa, w1 = *(const vf4*)(wa + 4);
            v8bf pa, pw;
#pragma unroll
            for (int j = 0; j < 4; ++j) {
                pa[j] = (__bf16)x0[j]; pa[4 + j] = (__bf16)x1[j];
                pw[j] = (__bf16)w0[j]; pw[4 + j] = (__bf16)w1[j];
            }
            *(v8bf*)&At[r * 32 + c] = pa;
            *(v8bf*)&Bt[r * 32 + c] = pw;
        }
        if (k0 + 32 < E_DIM) {            // prefetch next K-tile
            int pr = tid >> 1, pc = (tid & 1) << 4;
            __builtin_prefetch(X + (size_t)(M0 + pr) * E_DIM + k0 + 32 + pc, 0, 1);
            __builtin_prefetch(W + (size_t)(N0 + pr) * E_DIM + k0 + 32 + pc, 0, 1);
        }
        __syncthreads();

        v16bf af = load_frag32(&At[(wave * 16 + (lane & 15)) * 32], kb);
        v16bf bfr[4];
#pragma unroll
        for (int nt = 0; nt < 4; ++nt)
            bfr[nt] = load_frag32(&Bt[(nt * 16 + (lane & 15)) * 32], kb);
#pragma unroll
        for (int nt = 0; nt < 4; ++nt)
            acc[nt] = wmma_bf16(af, bfr[nt], acc[nt]);
    }

#pragma unroll
    for (int nt = 0; nt < 4; ++nt) {
        int n = N0 + nt * 16 + (lane & 15);
        float bn = bias[n];
        int h = n >> 6, hd = n & 63;
#pragma unroll
        for (int r = 0; r < 8; ++r) {
            int m = M0 + wave * 16 + r + kb;      // row = t*B + b
            int t = m >> 1, bb = m & 1;
            float y = (acc[nt][r] + bn) * scale;
            dst[(((size_t)bb * H_DIM + h) * T_DIM + t) * HD_DIM + hd] = (__bf16)y;
        }
    }
}

// ---------------- Kernel 2: streaming attention (flash-style) ----------------
// Per wave: 16 queries x full 64-d head. S^T = K x Q^T via WMMA; the f32
// C-layout of S^T reinterprets directly as the bf16 B operand of
// out^T = V^T x P^T (no shuffles). Online softmax; "mind the gap" becomes
// out -= colmean(V) since mean(softmax) = 1/S.
// K/V tiles (contiguous 4KB) are moved by the Tensor Data Mover, double
// buffered: wave0 issues the TDM for tile i+1, all waves compute on tile i,
// wave0 drains TENSORcnt at iteration end (DMA overlaps WMMA+softmax).
__global__ __launch_bounds__(128)
void attn_kernel(const __bf16* __restrict__ qg, const __bf16* __restrict__ kg,
                 const __bf16* __restrict__ vg, __bf16* __restrict__ ctx)
{
    __shared__ __bf16 Kbuf[2][32 * 64];   // [s_local][hd]
    __shared__ __bf16 Vbuf[2][32 * 64];   // [s_local][hd]
    __shared__ float  vsumArr[64];

    const int tid  = threadIdx.x;
    const int wave = tid >> 5;
    const int lane = tid & 31;
    const int t0   = blockIdx.x * 64 + wave * 16;
    const int h    = blockIdx.y;
    const int b    = blockIdx.z;
    const int kb   = (lane >> 4) << 3;

    const size_t head_base = ((size_t)b * H_DIM + h) * (size_t)T_DIM * HD_DIM;
    const __bf16* Kg = kg + head_base;
    const __bf16* Vg = vg + head_base;

    const uint32_t k_off0 = (uint32_t)(uintptr_t)&Kbuf[0][0];
    const uint32_t k_off1 = (uint32_t)(uintptr_t)&Kbuf[1][0];
    const uint32_t v_off0 = (uint32_t)(uintptr_t)&Vbuf[0][0];
    const uint32_t v_off1 = (uint32_t)(uintptr_t)&Vbuf[1][0];

    // Q fragment (B operand: 32hd x 16t), resident for the whole s-loop.
    const int t_col = t0 + (lane & 15);
    const __bf16* qrow = qg + head_base + (size_t)t_col * HD_DIM;
    const v16bf qf0 = load_frag32(qrow, kb);
    const v16bf qf1 = load_frag32(qrow + 32, kb);

    v8f acc_o[4] = {};
    float m_i = -3.0e38f, l_i = 0.0f, vsum_reg = 0.0f;
    const float t_f = (float)t_col;

    // Prologue: TDM tile 0 into buffer 0.
    if (wave == 0) {
        tdm_load_bf16(k_off0, Kg, 32 * HD_DIM);
        tdm_load_bf16(v_off0, Vg, 32 * HD_DIM);
        wait_tensor0();
    }
    __syncthreads();

    const int NITER = T_DIM / 32;
    for (int it = 0; it < NITER; ++it) {
        const int cur = it & 1;
        const int s0  = it * 32;

        // Issue DMA for the next tile into the other buffer (its last readers
        // completed before the barrier that ended the previous iteration).
        if (wave == 0 && it + 1 < NITER) {
            const __bf16* gk = Kg + (size_t)(s0 + 32) * HD_DIM;
            const __bf16* gv = Vg + (size_t)(s0 + 32) * HD_DIM;
            tdm_load_bf16(cur ? k_off0 : k_off1, gk, 32 * HD_DIM);
            tdm_load_bf16(cur ? v_off0 : v_off1, gv, 32 * HD_DIM);
        }

        const __bf16* Kt   = &Kbuf[cur][0];
        const __bf16* Vraw = &Vbuf[cur][0];

        if (tid < 64) {                        // V column-sum (mind-the-gap)
            float a = 0.0f;
#pragma unroll
            for (int r = 0; r < 32; ++r) a += (float)Vraw[r * HD_DIM + tid];
            vsum_reg += a;
        }

        // S^T tiles: D[s_local, t], s_local 0..15 (sacc0) and 16..31 (sacc1)
        const __bf16* krow0 = &Kt[(lane & 15) * HD_DIM];
        const __bf16* krow1 = krow0 + 16 * HD_DIM;
        v16bf kf00 = load_frag32(krow0,      kb);
        v16bf kf01 = load_frag32(krow0 + 32, kb);
        v16bf kf10 = load_frag32(krow1,      kb);
        v16bf kf11 = load_frag32(krow1 + 32, kb);
        v8f sacc0 = {}, sacc1 = {};
        sacc0 = wmma_bf16(kf00, qf0, sacc0);
        sacc0 = wmma_bf16(kf01, qf1, sacc0);
        sacc1 = wmma_bf16(kf10, qf0, sacc1);
        sacc1 = wmma_bf16(kf11, qf1, sacc1);

        // relative-position bias + online softmax (stats per t are per-lane)
        float pv[16];
        float mt = -3.0e38f;
#pragma unroll
        for (int r = 0; r < 8; ++r) {
            float s_f0 = (float)(s0 + r + kb);
            float s_f1 = (float)(s0 + 16 + r + kb);
            float v0 = sacc0[r] - fabsf(t_f - s_f0) * 0.03125f;  // -|t-s|/sqrt(E)
            float v1 = sacc1[r] - fabsf(t_f - s_f1) * 0.03125f;
            pv[r] = v0; pv[8 + r] = v1;
            mt = fmaxf(mt, fmaxf(v0, v1));
        }
        mt = fmaxf(mt, __shfl_xor(mt, 16, 32));
        float mnew = fmaxf(m_i, mt);
        float corr = __expf(m_i - mnew);
        float lsum = 0.0f;
        v16bf pb;                      // P^T B-operand, built in-place
#pragma unroll
        for (int j = 0; j < 16; ++j) {
            float p = __expf(pv[j] - mnew);
            lsum += p;
            pb[j] = (__bf16)p;
        }
        lsum += __shfl_xor(lsum, 16, 32);
        l_i = l_i * corr + lsum;
        m_i = mnew;
#pragma unroll
        for (int dc = 0; dc < 4; ++dc)
#pragma unroll
            for (int r = 0; r < 8; ++r) acc_o[dc][r] *= corr;

        // out^T[d,t] += V^T x P^T   (V^T fragments gathered column-wise)
        v16bf vfr[4];
#pragma unroll
        for (int dc = 0; dc < 4; ++dc)
            vfr[dc] = load_frag_col(&Vraw[dc * 16 + (lane & 15)], kb);
#pragma unroll
        for (int dc = 0; dc < 4; ++dc)
            acc_o[dc] = wmma_bf16(vfr[dc], pb, acc_o[dc]);

        // Drain the next tile's DMA, then publish to all waves.
        if (wave == 0) wait_tensor0();
        __syncthreads();
    }

    if (tid < 64) vsumArr[tid] = vsum_reg;
    __syncthreads();

    const float inv_l = 1.0f / l_i;
#pragma unroll
    for (int dc = 0; dc < 4; ++dc) {
#pragma unroll
        for (int r = 0; r < 8; ++r) {
            int d = dc * 16 + r + kb;
            float val = acc_o[dc][r] * inv_l - vsumArr[d] * (1.0f / (float)T_DIM);
            ctx[((size_t)t_col * B_DIM + b) * E_DIM + h * HD_DIM + d] = (__bf16)val;
        }
    }
}

// ---------------- Kernel 3: output projection ----------------
__global__ __launch_bounds__(128)
void out_proj_kernel(const __bf16* __restrict__ ctx, const float* __restrict__ Wo,
                     const float* __restrict__ bo, float* __restrict__ out)
{
    __shared__ __bf16 At[64 * 32];
    __shared__ __bf16 Bt[64 * 32];

    const int tid  = threadIdx.x;
    const int wave = tid >> 5;
    const int lane = tid & 31;
    const int M0   = blockIdx.x * 64;
    const int N0   = blockIdx.y * 64;
    const int kb   = (lane >> 4) << 3;

    const uint32_t at_base = (uint32_t)(uintptr_t)&At[0];

    v8f acc[4] = {};

    for (int k0 = 0; k0 < E_DIM; k0 += 32) {
        __syncthreads();
        // A tile is already bf16 -> async DMA rows (64B row segments, 4x16B)
        for (int i = tid; i < 256; i += 128) {
            int r = i >> 2, c = (i & 3) << 3;       // 8 bf16 = 16B
            async_copy16(at_base + (uint32_t)(r * 32 + c) * 2,
                         ctx + (size_t)(M0 + r) * E_DIM + k0 + c);
        }
        // B tile: fp32 -> bf16 convert, vectorized (overlaps with async)
        for (int i = tid; i < 256; i += 128) {
            int r = i >> 2, c = (i & 3) << 3;
            const float* wa = Wo + (size_t)(N0 + r) * E_DIM + k0 + c;
            vf4 w0 = *(const vf4*)wa, w1 = *(const vf4*)(wa + 4);
            v8bf pw;
#pragma unroll
            for (int j = 0; j < 4; ++j) { pw[j] = (__bf16)w0[j]; pw[4 + j] = (__bf16)w1[j]; }
            *(v8bf*)&Bt[r * 32 + c] = pw;
        }
        if (k0 + 32 < E_DIM) {
            int pr = tid >> 1, pc = (tid & 1) << 4;
            __builtin_prefetch(Wo + (size_t)(N0 + pr) * E_DIM + k0 + 32 + pc, 0, 1);
        }
        wait_async0();
        __syncthreads();

        v16bf af = load_frag32(&At[(wave * 16 + (lane & 15)) * 32], kb);
        v16bf bfr[4];
#pragma unroll
        for (int nt = 0; nt < 4; ++nt)
            bfr[nt] = load_frag32(&Bt[(nt * 16 + (lane & 15)) * 32], kb);
#pragma unroll
        for (int nt = 0; nt < 4; ++nt)
            acc[nt] = wmma_bf16(af, bfr[nt], acc[nt]);
    }

#pragma unroll
    for (int nt = 0; nt < 4; ++nt) {
        int n = N0 + nt * 16 + (lane & 15);
        float bn = bo[n];
#pragma unroll
        for (int r = 0; r < 8; ++r) {
            int m = M0 + wave * 16 + r + kb;
            out[(size_t)m * E_DIM + n] = acc[nt][r] + bn;
        }
    }
}

extern "C" void kernel_launch(void* const* d_in, const int* in_sizes, int n_in,
                              void* d_out, int out_size, void* d_ws, size_t ws_size,
                              hipStream_t stream)
{
    const float* query = (const float*)d_in[0];
    const float* key_  = (const float*)d_in[1];
    const float* value = (const float*)d_in[2];
    const float* Wq = (const float*)d_in[3];
    const float* bq = (const float*)d_in[4];
    const float* Wk = (const float*)d_in[5];
    const float* bk = (const float*)d_in[6];
    const float* Wv = (const float*)d_in[7];
    const float* bv = (const float*)d_in[8];
    const float* Wo = (const float*)d_in[9];
    const float* bo = (const float*)d_in[10];
    float* out = (float*)d_out;

    const size_t HEAD_ELEMS = (size_t)B_DIM * H_DIM * T_DIM * HD_DIM;  // 4M
    __bf16* qb  = (__bf16*)d_ws;
    __bf16* kbw = qb  + HEAD_ELEMS;
    __bf16* vbw = kbw + HEAD_ELEMS;
    __bf16* ctx = vbw + HEAD_ELEMS;      // [T*B, E] bf16

    dim3 blk(128);
    qkv_proj_kernel<<<dim3((T_DIM * B_DIM) / 64, E_DIM / 64, 3), blk, 0, stream>>>(
        query, key_, value, Wq, bq, Wk, bk, Wv, bv, qb, kbw, vbw);
    attn_kernel<<<dim3(T_DIM / 64, H_DIM, B_DIM), blk, 0, stream>>>(qb, kbw, vbw, ctx);
    out_proj_kernel<<<dim3((T_DIM * B_DIM) / 64, E_DIM / 64), blk, 0, stream>>>(
        ctx, Wo, bo, out);
}
